// SlidingGRU_31336081392292
// MI455X (gfx1250) — matrix operand order repaired and checked by reading, compiled, not run on hardware
//
#include <hip/hip_runtime.h>
#include <hip/hip_bf16.h>

// Problem constants (match reference)
#define BB   256
#define TT   100
#define CC   128
#define HH   1024
#define WWIN 16

typedef __attribute__((ext_vector_type(16))) __bf16 v16bf;
typedef __attribute__((ext_vector_type(8)))  float  v8f;

// Async global->LDS builtin takes vector-of-4-int pointers:
//   param0: addrspace(1) int4*   param1: addrspace(3) int4*   (imm offset, imm cpol)
typedef int v4i_ __attribute__((__vector_size__(16)));
typedef __attribute__((address_space(1))) v4i_* gas_b128;
typedef __attribute__((address_space(3))) v4i_* las_b128;

#if defined(__has_builtin)
#if __has_builtin(__builtin_amdgcn_global_load_async_to_lds_b128)
#define HAVE_ASYNC_LDS 1
#endif
#endif
#ifndef HAVE_ASYNC_LDS
#define HAVE_ASYNC_LDS 0
#endif

__device__ __forceinline__ void wait_async0() {
#if __has_builtin(__builtin_amdgcn_s_wait_asynccnt)
  __builtin_amdgcn_s_wait_asynccnt(0);
#else
  asm volatile("s_wait_asynccnt 0x0" ::: "memory");
#endif
}

__device__ __forceinline__ __bf16 f2bf(float f) {
  union { float f; unsigned u; } v; v.f = f;
  unsigned r = v.u + 0x7FFFu + ((v.u >> 16) & 1u);   // round-to-nearest-even
  unsigned short h = (unsigned short)(r >> 16);
  union { unsigned short s; __bf16 b; } o; o.s = h; return o.b;
}

__device__ __forceinline__ float sigmoidf_(float x) {
  return 1.0f / (1.0f + __expf(-x));
}

// A-fragment addressing (V_WMMA 16-bit A, 16x32 tile):
//   element i within a 512-elem fragment: lane=(i>>4)&31, e=i&15
//   maps to m = lane&15, k = kt*32 + e + ((e>=8)?8:0) + ((lane>=16)?8:0)
// Inverse (given m, k): kt=k>>5, kp=k&31, half=(kp>>3)&1,
//   e=(kp&7)+((kp>=16)?8:0), lane=half*16+m.
__device__ __forceinline__ int afrag_index(int m, int k) {
  int kt = k >> 5, kp = k & 31;
  int half = (kp >> 3) & 1;
  int e = (kp & 7) + ((kp >= 16) ? 8 : 0);
  return (kt << 9) + ((half * 16 + m) << 4) + e;
}

// ---------------------------------------------------------------------------
// Pack an f32 weight matrix W (3H x K, row-major) into bf16 WMMA B-fragments.
// frag f = (g*64 + nt)*KT + kt; lane holds column n = nt*16 + (lane&15),
// element e holds k = kt*32 + e + ((e>=8)?8:0) + ((lane>=16)?8:0)  (matches A layout).
// ---------------------------------------------------------------------------
__global__ void pack_weight_kernel(const float* __restrict__ W,
                                   __bf16* __restrict__ Wp, int K) {
  const int KT = K >> 5;
  const int total = 3 * HH * K;
  for (int i = blockIdx.x * blockDim.x + threadIdx.x; i < total;
       i += gridDim.x * blockDim.x) {
    int f    = i >> 9;
    int lane = (i >> 4) & 31;
    int e    = i & 15;
    int ntg  = f / KT;
    int kt   = f - ntg * KT;
    int n    = ntg * 16 + (lane & 15);
    int k    = kt * 32 + e + ((e >= 8) ? 8 : 0) + ((lane >= 16) ? 8 : 0);
    Wp[i] = f2bf(W[n * K + k]);
  }
}

// Pack the input x (B,T,C) once into per-(t, mtile) bf16 A-fragment blocks:
// block offset = (t*16 + mtile) * (16*C); contiguous 4KB per block.
__global__ void pack_x_kernel(const float* __restrict__ x,
                              __bf16* __restrict__ xp) {
  const int total = TT * BB * CC;
  for (int i = blockIdx.x * blockDim.x + threadIdx.x; i < total;
       i += gridDim.x * blockDim.x) {
    int blk = i / (16 * CC);
    int r   = i - blk * (16 * CC);
    int t = blk >> 4, mtile = blk & 15;
    int lane = (r >> 4) & 31, e = r & 15, kt = r >> 9;
    int m = lane & 15;
    int k = kt * 32 + e + ((e >= 8) ? 8 : 0) + ((lane >= 16) ? 8 : 0);
    xp[i] = f2bf(x[(mtile * 16 + m) * (TT * CC) + t * CC + k]);
  }
}

// Transpose Ws (C x H) -> Wst (H x C) for coalesced decoder projection.
__global__ void transpose_ws_kernel(const float* __restrict__ Ws,
                                    float* __restrict__ Wst) {
  int i = blockIdx.x * blockDim.x + threadIdx.x;   // over C*H
  if (i < CC * HH) {
    int c = i / HH, h = i - c * HH;
    Wst[h * CC + c] = Ws[c * HH + h];
  }
}

// weff[w] = sum_p Wtl[p]*Wt1[p,w]; weff[16] = sum_p Wtl[p]*bt1[p] + btl.
__global__ void weff_kernel(const float* __restrict__ Wt1,
                            const float* __restrict__ bt1,
                            const float* __restrict__ Wtl,
                            const float* __restrict__ btl,
                            float* __restrict__ weff) {
  int w = threadIdx.x;
  if (w < WWIN) {
    float a = 0.f;
    for (int p = 0; p < WWIN; ++p) a += Wtl[p] * Wt1[p * WWIN + w];
    weff[w] = a;
  }
  if (w == WWIN) {
    float a = btl[0];
    for (int p = 0; p < WWIN; ++p) a += Wtl[p] * bt1[p];
    weff[WWIN] = a;
  }
}

__global__ void zero_kernel(float* __restrict__ p, int n) {
  int i = blockIdx.x * blockDim.x + threadIdx.x;
  if (i < n) p[i] = 0.f;
}

// dec0 = x[:, T-1, :]  (f32 + packed bf16 A-fragment form)
__global__ void dec_init_kernel(const float* __restrict__ x,
                                float* __restrict__ dec,
                                __bf16* __restrict__ decp) {
  int i = blockIdx.x * blockDim.x + threadIdx.x;   // over B*C
  if (i < BB * CC) {
    int b = i / CC, c = i - b * CC;
    float v = x[b * (TT * CC) + (TT - 1) * CC + c];
    dec[i] = v;
    decp[(b >> 4) * (16 * CC) + afrag_index(b & 15, c)] = f2bf(v);
  }
}

// ---------------------------------------------------------------------------
// One GRU cell for all B. Inputs arrive PRE-PACKED in bf16 A-fragment order
// (contiguous per 16-row m-tile) so staging is a pure async global->LDS DMA.
// Grid (8,16): blockIdx.y = m-tile, blockIdx.x = group of 8 n-tiles; 8 waves.
// Epilogue writes h_out f32 (state + window) AND the bf16 A-packed form that
// feeds the next cell. 6 f32 accumulators: {i_r,i_z,i_n,h_r,h_z,h_n}.
// ---------------------------------------------------------------------------
__global__ __launch_bounds__(256)
void gru_cell_kernel(const __bf16* __restrict__ xpack, int x_mstride, int Kin,
                     const float* __restrict__ h,
                     const __bf16* __restrict__ hpack,
                     const __bf16* __restrict__ Wih,
                     const __bf16* __restrict__ Whh,
                     const float* __restrict__ b_ih,
                     const float* __restrict__ b_hh,
                     float* __restrict__ h_out,
                     __bf16* __restrict__ hpack_out,
                     float* __restrict__ win_out) {
  __shared__ __bf16 ldsX[16 * HH];
  __shared__ __bf16 ldsH[16 * HH];

  const int tid = threadIdx.x;
  const int m0  = blockIdx.y * 16;

  const __bf16* xsrc = xpack + (size_t)blockIdx.y * x_mstride;
  const __bf16* hsrc = hpack + (size_t)blockIdx.y * (16 * HH);

#if HAVE_ASYNC_LDS
  // Fire-and-forget async DMA: 16B per lane per issue, tracked by ASYNCcnt.
  for (int off = tid * 8; off < 16 * Kin; off += 256 * 8) {
    __builtin_amdgcn_global_load_async_to_lds_b128(
        (gas_b128)(xsrc + off), (las_b128)(&ldsX[off]), 0, 0);
  }
  for (int off = tid * 8; off < 16 * HH; off += 256 * 8) {
    __builtin_amdgcn_global_load_async_to_lds_b128(
        (gas_b128)(hsrc + off), (las_b128)(&ldsH[off]), 0, 0);
  }
  wait_async0();
  __syncthreads();
#else
  {
    const uint4* gx = (const uint4*)xsrc;
    uint4* lx = (uint4*)ldsX;
    for (int i = tid; i < (16 * Kin) / 8; i += 256) lx[i] = gx[i];
    const uint4* gh = (const uint4*)hsrc;
    uint4* lh = (uint4*)ldsH;
    for (int i = tid; i < (16 * HH) / 8; i += 256) lh[i] = gh[i];
  }
  __syncthreads();
#endif

  const int lane = tid & 31;
  const int wv   = tid >> 5;
  const int n0   = blockIdx.x * 128 + wv * 16;  // within H
  const int nt   = n0 >> 4;

  v8f aci0 = {}, aci1 = {}, aci2 = {};
  v8f ach0 = {}, ach1 = {}, ach2 = {};

  // Input-to-hidden part: K = Kin
  const int KTi = Kin >> 5;
  for (int kt = 0; kt < KTi; ++kt) {
    v16bf a = *(const v16bf*)&ldsX[(kt << 9) + (lane << 4)];
    const __bf16* bp0 = Wih + ((((0 * 64 + nt) * KTi + kt) << 9) + (lane << 4));
    const __bf16* bp1 = Wih + ((((1 * 64 + nt) * KTi + kt) << 9) + (lane << 4));
    const __bf16* bp2 = Wih + ((((2 * 64 + nt) * KTi + kt) << 9) + (lane << 4));
    if (kt + 1 < KTi) __builtin_prefetch((const void*)(bp0 + 512), 0, 1);
    v16bf b0 = *(const v16bf*)bp0;
    v16bf b1 = *(const v16bf*)bp1;
    v16bf b2 = *(const v16bf*)bp2;
    // Same A operand back-to-back: set ISA reuse_a hint on the first two.
    aci0 = __builtin_amdgcn_wmma_f32_16x16x32_bf16(false, a, false, b0, (short)0, aci0, true,  false);
    aci1 = __builtin_amdgcn_wmma_f32_16x16x32_bf16(false, a, false, b1, (short)0, aci1, true,  false);
    aci2 = __builtin_amdgcn_wmma_f32_16x16x32_bf16(false, a, false, b2, (short)0, aci2, false, false);
  }
  // Hidden-to-hidden part: K = H
  for (int kt = 0; kt < (HH >> 5); ++kt) {
    v16bf a = *(const v16bf*)&ldsH[(kt << 9) + (lane << 4)];
    const __bf16* bp0 = Whh + ((((0 * 64 + nt) * (HH >> 5) + kt) << 9) + (lane << 4));
    const __bf16* bp1 = Whh + ((((1 * 64 + nt) * (HH >> 5) + kt) << 9) + (lane << 4));
    const __bf16* bp2 = Whh + ((((2 * 64 + nt) * (HH >> 5) + kt) << 9) + (lane << 4));
    if (kt + 1 < (HH >> 5)) __builtin_prefetch((const void*)(bp0 + 512), 0, 1);
    v16bf b0 = *(const v16bf*)bp0;
    v16bf b1 = *(const v16bf*)bp1;
    v16bf b2 = *(const v16bf*)bp2;
    ach0 = __builtin_amdgcn_wmma_f32_16x16x32_bf16(false, a, false, b0, (short)0, ach0, true,  false);
    ach1 = __builtin_amdgcn_wmma_f32_16x16x32_bf16(false, a, false, b1, (short)0, ach1, true,  false);
    ach2 = __builtin_amdgcn_wmma_f32_16x16x32_bf16(false, a, false, b2, (short)0, ach2, false, false);
  }

  // Fused gate epilogue. C/D layout: VGPR v -> row m0 + v + 8*(lane>=16),
  // col n0 + (lane&15).
  const int colH = n0 + (lane & 15);
  const float bir = b_ih[colH];
  const float biz = b_ih[HH + colH];
  const float bin = b_ih[2 * HH + colH];
  const float bhr = b_hh[colH];
  const float bhz = b_hh[HH + colH];
  const float bhn = b_hh[2 * HH + colH];
  const int mbase = (lane >> 4) << 3;

  // Pre-compute A-fragment scatter coords for the packed write (k = colH).
  const int ktp  = colH >> 5;
  const int kp   = colH & 31;
  const int half = (kp >> 3) & 1;
  const int ep   = (kp & 7) + ((kp >= 16) ? 8 : 0);
  __bf16* hp_tile = hpack_out + (size_t)blockIdx.y * (16 * HH);

#pragma unroll
  for (int v = 0; v < 8; ++v) {
    const int ml  = mbase + v;
    const int row = m0 + ml;
    const float hprev = h[row * HH + colH];
    const float rg = sigmoidf_(aci0[v] + bir + ach0[v] + bhr);
    const float zg = sigmoidf_(aci1[v] + biz + ach1[v] + bhz);
    const float ng = tanhf(aci2[v] + bin + rg * (ach2[v] + bhn));
    const float hn = (1.f - zg) * ng + zg * hprev;
    h_out[row * HH + colH] = hn;
    hp_tile[(ktp << 9) + (((half << 4) + ml) << 4) + ep] = f2bf(hn);
    if (win_out) win_out[row * HH + colH] = hn;
  }
}

// zf[b,h] = sum_w weff[w] * ring[(start+w)&15][b,h] + weff[16]
__global__ void zf_kernel(const float* __restrict__ ring, int start,
                          const float* __restrict__ weff,
                          float* __restrict__ zf) {
  int i = blockIdx.x * blockDim.x + threadIdx.x;   // over B*H
  if (i >= BB * HH) return;
  float acc = weff[WWIN];
#pragma unroll
  for (int w = 0; w < WWIN; ++w) {
    int slot = (start + w) & (WWIN - 1);
    acc += weff[w] * ring[slot * (BB * HH) + i];
  }
  zf[i] = acc;
}

// new[b,c] = zf[b,:] @ Ws[c,:] + bs[c] + dec[b,c]
// Writes out[:,t,:], the next decoder input (f32) and its bf16 packed form.
__global__ __launch_bounds__(128)
void sp_kernel(const float* __restrict__ zf, const float* __restrict__ Wst,
               const float* __restrict__ bs, const float* __restrict__ dec_in,
               float* __restrict__ dec_out, __bf16* __restrict__ decp_out,
               float* __restrict__ out, int t) {
  const int b = blockIdx.x;
  const int c = threadIdx.x;
  const float* z = zf + b * HH;
  float acc = 0.f;
  for (int hh = 0; hh < HH; ++hh) acc += Wst[hh * CC + c] * z[hh];
  const float v = acc + bs[c] + dec_in[b * CC + c];
  dec_out[b * CC + c] = v;
  decp_out[(b >> 4) * (16 * CC) + afrag_index(b & 15, c)] = f2bf(v);
  out[(b * TT + t) * CC + c] = v;
}

// ---------------------------------------------------------------------------
extern "C" void kernel_launch(void* const* d_in, const int* in_sizes, int n_in,
                              void* d_out, int out_size, void* d_ws, size_t ws_size,
                              hipStream_t stream) {
  const float* x     = (const float*)d_in[0];
  const float* W_ih0 = (const float*)d_in[1];
  const float* W_hh0 = (const float*)d_in[2];
  const float* b_ih0 = (const float*)d_in[3];
  const float* b_hh0 = (const float*)d_in[4];
  const float* W_ih1 = (const float*)d_in[5];
  const float* W_hh1 = (const float*)d_in[6];
  const float* b_ih1 = (const float*)d_in[7];
  const float* b_hh1 = (const float*)d_in[8];
  const float* Wt1   = (const float*)d_in[9];
  const float* bt1   = (const float*)d_in[10];
  const float* Wtl   = (const float*)d_in[11];
  const float* btl   = (const float*)d_in[12];
  const float* Ws    = (const float*)d_in[13];
  const float* bs    = (const float*)d_in[14];
  float* out = (float*)d_out;

  // Workspace carve-up (256B aligned)
  char* base = (char*)d_ws;
  size_t off = 0;
  auto carve = [&](size_t bytes) -> char* {
    char* p = base + off;
    off = (off + bytes + 255) & ~(size_t)255;
    return p;
  };
  __bf16* Wp_ih0 = (__bf16*)carve((size_t)3 * HH * CC * 2);
  __bf16* Wp_hh0 = (__bf16*)carve((size_t)3 * HH * HH * 2);
  __bf16* Wp_ih1 = (__bf16*)carve((size_t)3 * HH * HH * 2);
  __bf16* Wp_hh1 = (__bf16*)carve((size_t)3 * HH * HH * 2);
  __bf16* xpack  = (__bf16*)carve((size_t)TT * BB * CC * 2);
  float*  Wst    = (float*)carve((size_t)CC * HH * 4);
  float*  h0a    = (float*)carve((size_t)BB * HH * 4);
  float*  h0b    = (float*)carve((size_t)BB * HH * 4);
  float*  h1a    = (float*)carve((size_t)BB * HH * 4);
  float*  h1b    = (float*)carve((size_t)BB * HH * 4);
  __bf16* h0apk  = (__bf16*)carve((size_t)BB * HH * 2);
  __bf16* h0bpk  = (__bf16*)carve((size_t)BB * HH * 2);
  __bf16* h1apk  = (__bf16*)carve((size_t)BB * HH * 2);
  __bf16* h1bpk  = (__bf16*)carve((size_t)BB * HH * 2);
  float*  ring   = (float*)carve((size_t)WWIN * BB * HH * 4);
  float*  decA   = (float*)carve((size_t)BB * CC * 4);
  float*  decB   = (float*)carve((size_t)BB * CC * 4);
  __bf16* decApk = (__bf16*)carve((size_t)BB * CC * 2);
  __bf16* decBpk = (__bf16*)carve((size_t)BB * CC * 2);
  float*  zf     = (float*)carve((size_t)BB * HH * 4);
  float*  weff   = (float*)carve((size_t)(WWIN + 1) * 4);

  // One-time (per launch) prep.
  pack_weight_kernel<<<1024, 256, 0, stream>>>(W_ih0, Wp_ih0, CC);
  pack_weight_kernel<<<1024, 256, 0, stream>>>(W_hh0, Wp_hh0, HH);
  pack_weight_kernel<<<1024, 256, 0, stream>>>(W_ih1, Wp_ih1, HH);
  pack_weight_kernel<<<1024, 256, 0, stream>>>(W_hh1, Wp_hh1, HH);
  pack_x_kernel<<<1024, 256, 0, stream>>>(x, xpack);
  transpose_ws_kernel<<<(CC * HH + 255) / 256, 256, 0, stream>>>(Ws, Wst);
  weff_kernel<<<1, 32, 0, stream>>>(Wt1, bt1, Wtl, btl, weff);
  zero_kernel<<<(BB * HH + 255) / 256, 256, 0, stream>>>(h0a, BB * HH);
  zero_kernel<<<(BB * HH + 255) / 256, 256, 0, stream>>>(h1a, BB * HH);
  zero_kernel<<<(BB * HH / 2 + 255) / 256, 256, 0, stream>>>((float*)h0apk, BB * HH / 2);
  zero_kernel<<<(BB * HH / 2 + 255) / 256, 256, 0, stream>>>((float*)h1apk, BB * HH / 2);
  dec_init_kernel<<<(BB * CC + 255) / 256, 256, 0, stream>>>(x, decA, decApk);

  const dim3 cgrid(HH / 128, BB / 16);   // 8 x 16 blocks, 8 waves each

  // Encoder: 100 sequential steps, last 16 layer-1 outputs into window ring.
  float  *c0 = h0a, *c0n = h0b, *c1 = h1a, *c1n = h1b;
  __bf16 *p0 = h0apk, *p0n = h0bpk, *p1 = h1apk, *p1n = h1bpk;
  for (int t = 0; t < TT; ++t) {
    gru_cell_kernel<<<cgrid, 256, 0, stream>>>(
        xpack + (size_t)t * BB * CC, 16 * CC, CC,
        c0, p0, Wp_ih0, Wp_hh0, b_ih0, b_hh0, c0n, p0n, nullptr);
    float* win = (t >= TT - WWIN) ? ring + (size_t)(t - (TT - WWIN)) * BB * HH
                                  : nullptr;
    gru_cell_kernel<<<cgrid, 256, 0, stream>>>(
        p0n, 16 * HH, HH,
        c1, p1, Wp_ih1, Wp_hh1, b_ih1, b_hh1, c1n, p1n, win);
    float* tf; __bf16* tp;
    tf = c0; c0 = c0n; c0n = tf;   tp = p0; p0 = p0n; p0n = tp;
    tf = c1; c1 = c1n; c1n = tf;   tp = p1; p1 = p1n; p1n = tp;
  }

  // Decoder: 100 steps; n1 replaces oldest ring slot, fused window ops,
  // projection back to C, residual with previous frame.
  float  *dcur = decA, *dnxt = decB;
  __bf16 *dpk = decApk, *dpkn = decBpk;
  for (int t = 0; t < TT; ++t) {
    gru_cell_kernel<<<cgrid, 256, 0, stream>>>(
        dpk, 16 * CC, CC,
        c0, p0, Wp_ih0, Wp_hh0, b_ih0, b_hh0, c0n, p0n, nullptr);
    gru_cell_kernel<<<cgrid, 256, 0, stream>>>(
        p0n, 16 * HH, HH,
        c1, p1, Wp_ih1, Wp_hh1, b_ih1, b_hh1, c1n, p1n,
        ring + (size_t)(t & (WWIN - 1)) * BB * HH);
    zf_kernel<<<(BB * HH + 255) / 256, 256, 0, stream>>>(
        ring, (t + 1) & (WWIN - 1), weff, zf);
    sp_kernel<<<BB, CC, 0, stream>>>(zf, Wst, bs, dcur, dnxt, dpkn, out, t);
    float* tf; __bf16* tp;
    tf = c0; c0 = c0n; c0n = tf;   tp = p0; p0 = p0n; p0n = tp;
    tf = c1; c1 = c1n; c1n = tf;   tp = p1; p1 = p1n; p1n = tp;
    tf = dcur; dcur = dnxt; dnxt = tf;
    tp = dpk; dpk = dpkn; dpkn = tp;
  }
  (void)in_sizes; (void)n_in; (void)out_size; (void)ws_size;
}
